// OptimizedMOEImproved_36197984371397
// MI455X (gfx1250) — compile-verified, matrix operand order checked by reading, not run on hardware
//
#include <hip/hip_runtime.h>
#include <hip/hip_bf16.h>

// ---------------------------------------------------------------------------
// Problem constants (from reference)
// ---------------------------------------------------------------------------
#define BATCH   64
#define CDIM    64      // input channels (= GEMM K)
#define ODIM    64      // output channels (= GEMM M)
#define HWDIM   6400    // 80*80 spatial (= GEMM N total)
#define NEXP    4
#define COLS_PER_BLOCK  128          // spatial columns per tile
#define TILES_PER_BLOCK 5            // 6400 = 10 blocks * 5 tiles * 128 cols
#define GRID_X (HWDIM / (COLS_PER_BLOCK * TILES_PER_BLOCK))   // 10

typedef __attribute__((ext_vector_type(16))) __bf16 v16bf;
typedef __attribute__((ext_vector_type(8)))  __bf16 v8bf;
typedef __attribute__((ext_vector_type(8)))  float  v8f;

// round-to-nearest-even f32 -> bf16 (bit ops; no reliance on cast support)
__device__ __forceinline__ unsigned bf16_bits(float f) {
    unsigned u = __builtin_bit_cast(unsigned, f);
    u += 0x7FFFu + ((u >> 16) & 1u);
    return u;                         // result in bits [31:16]
}
// convert two floats, store as one packed 32-bit LDS word (dst 4B-aligned)
__device__ __forceinline__ void st_bf16x2(__bf16* dst, float a, float b) {
    unsigned pk = (bf16_bits(a) >> 16) | (bf16_bits(b) & 0xFFFF0000u);
    *(unsigned*)dst = pk;
}

// SiLU with hardware reciprocal (v_rcp_f32) instead of IEEE divide sequence
__device__ __forceinline__ float silu(float v) {
    return v * __builtin_amdgcn_rcpf(1.0f + __expf(-v));
}

__device__ __forceinline__ v16bf ld_frag(const __bf16* p) {
    v8bf lo = *(const v8bf*)p;
    v8bf hi = *(const v8bf*)(p + 16);
    return __builtin_shufflevector(lo, hi,
               0, 1, 2, 3, 4, 5, 6, 7, 8, 9, 10, 11, 12, 13, 14, 15);
}

// ---------------------------------------------------------------------------
// Kernel 1: global-avg-pool -> router linear -> softmax -> top-2 (renorm)
// ---------------------------------------------------------------------------
__global__ __launch_bounds__(256)
void moe_router_kernel(const float* __restrict__ x,
                       const float* __restrict__ router_w,
                       const float* __restrict__ router_b,
                       int* __restrict__ idx_out,       // [B][2]
                       float* __restrict__ w_out) {     // [B][2]
    const int b    = blockIdx.x;
    const int tid  = threadIdx.x;
    const int wave = tid >> 5;
    const int lane = tid & 31;

    __shared__ float pooled[CDIM];
    __shared__ float logits[NEXP];

    const float* xb = x + (size_t)b * CDIM * HWDIM;

    #pragma unroll
    for (int cl = 0; cl < 8; ++cl) {
        const int ch = wave * 8 + cl;
        const float* p = xb + (size_t)ch * HWDIM;
        float s = 0.0f;
        for (int i = lane; i < HWDIM; i += 32) s += p[i];
        #pragma unroll
        for (int off = 16; off > 0; off >>= 1) s += __shfl_xor(s, off, 32);
        if (lane == 0) pooled[ch] = s * (1.0f / (float)HWDIM);
    }
    __syncthreads();

    if (tid < NEXP) {
        float acc = router_b[tid];
        #pragma unroll 8
        for (int c = 0; c < CDIM; ++c) acc += pooled[c] * router_w[tid * CDIM + c];
        logits[tid] = acc;
    }
    __syncthreads();

    if (tid == 0) {
        float m = logits[0];
        #pragma unroll
        for (int e = 1; e < NEXP; ++e) m = fmaxf(m, logits[e]);
        float p[NEXP], sum = 0.0f;
        #pragma unroll
        for (int e = 0; e < NEXP; ++e) { p[e] = __expf(logits[e] - m); sum += p[e]; }
        const float inv = 1.0f / sum;
        #pragma unroll
        for (int e = 0; e < NEXP; ++e) p[e] *= inv;

        int i0 = 0;
        #pragma unroll
        for (int e = 1; e < NEXP; ++e) if (p[e] > p[i0]) i0 = e;
        int i1 = (i0 == 0) ? 1 : 0;
        #pragma unroll
        for (int e = 0; e < NEXP; ++e) if (e != i0 && p[e] > p[i1]) i1 = e;

        const float s2 = 1.0f / (p[i0] + p[i1]);
        idx_out[b * 2 + 0] = i0;
        idx_out[b * 2 + 1] = i1;
        w_out[b * 2 + 0]   = p[i0] * s2;
        w_out[b * 2 + 1]   = p[i1] * s2;
    }
}

// ---------------------------------------------------------------------------
// Kernel 2: fused (shared + 2 gathered experts) 1x1-conv GEMM via bf16 WMMA.
// All 5 X tiles resident in LDS; row-blocks outer / tiles inner so each
// row-block's 6 A-fragments are loaded once and reused across 5 tiles.
// Grid: (10, 64). Block: 256 threads = 8 wave32. LDS: ~105.5 KB.
// ---------------------------------------------------------------------------
__global__ __launch_bounds__(256)
void moe_main_kernel(const float* __restrict__ x,
                     const float* __restrict__ shared_w,
                     const float* __restrict__ shared_gamma,
                     const float* __restrict__ shared_beta,
                     const float* __restrict__ expert_w,
                     const float* __restrict__ expert_gamma,
                     const float* __restrict__ expert_beta,
                     const int*   __restrict__ idx_r,
                     const float* __restrict__ w_r,
                     float* __restrict__ out) {
    const int b     = blockIdx.y;
    const int tile0 = blockIdx.x * TILES_PER_BLOCK;
    const int tid   = threadIdx.x;
    const int wv    = tid >> 5;
    const int lane  = tid & 31;

    // Wl first so A-fragment ds offsets fit the 16-bit immediate field.
    __shared__ __bf16 Wl[3 * ODIM * CDIM];                           // 24 KB
    __shared__ __bf16 Xs[TILES_PER_BLOCK][COLS_PER_BLOCK * CDIM];    // 80 KB
    __shared__ float  gmS[3 * ODIM];                                 // gamma*bn_scale
    __shared__ float  btS[3 * ODIM];

    const int   i0 = idx_r[b * 2 + 0];
    const int   i1 = idx_r[b * 2 + 1];
    const float w0 = w_r[b * 2 + 0];
    const float w1 = w_r[b * 2 + 1];

    // ---- stage weights (bf16, packed-pair stores): shared, expert[i0], expert[i1]
    for (int i = tid; i < (3 * ODIM * CDIM) / 2; i += 256) {
        const int e2 = i * 2;
        const int br = e2 >> 12;          // / 4096
        const int rk = e2 & 4095;
        const float* src = (br == 0) ? shared_w
                                     : expert_w + (size_t)((br == 1) ? i0 : i1) * ODIM * CDIM;
        st_bf16x2(&Wl[e2], src[rk], src[rk + 1]);
    }
    // ---- stage BN params, pre-scaled by 1/sqrt(1+eps) ----
    if (tid < 3 * ODIM) {
        const int br = tid >> 6, ch = tid & 63;
        const float bns = rsqrtf(1.0f + 1e-5f);
        float g, bt;
        if (br == 0) { g = shared_gamma[ch]; bt = shared_beta[ch]; }
        else {
            const int e = (br == 1) ? i0 : i1;
            g  = expert_gamma[e * ODIM + ch];
            bt = expert_beta [e * ODIM + ch];
        }
        gmS[tid] = g * bns;
        btS[tid] = bt;
    }

    // ---- stage ALL 5 X tiles: x[b][k][col] -> Xs[t][c][k] (bf16) ----
    // unit = (tile t, k-pair kp, c-pair cp): 2 float2 global loads ->
    // 2 packed 32-bit LDS stores. float2 along c keeps loads coalesced.
    const float* xb0 = x + (size_t)b * CDIM * HWDIM + tile0 * COLS_PER_BLOCK;
    for (int i = tid; i < TILES_PER_BLOCK * 32 * 64; i += 256) {
        const int t  = i >> 11;            // / 2048 units per tile
        const int r  = i & 2047;
        const int kp = r >> 6;             // 0..31  -> k = 2*kp
        const int cp = r & 63;             // 0..63  -> c = 2*cp
        const int k  = kp * 2;
        const int c  = cp * 2;
        const float* xt = xb0 + t * COLS_PER_BLOCK;
        const float2 lo = *(const float2*)&xt[(size_t)k * HWDIM + c];
        const float2 hi = *(const float2*)&xt[(size_t)(k + 1) * HWDIM + c];
        st_bf16x2(&Xs[t][c * CDIM + k],       lo.x, hi.x);
        st_bf16x2(&Xs[t][(c + 1) * CDIM + k], lo.y, hi.y);
    }
    __syncthreads();

    const int n16   = lane & 15;               // column within slab / N index
    const int khalf = (lane >> 4) << 3;        // 0 or 8: K sub-half per ISA layout
    const int mofs  = (lane >> 4) * 8;         // C/D layout: VGPR j -> rows j / j+8
    const int lcol  = wv * 16 + n16;           // this wave's column (local)

    // per-wave output base; all later stores use small structured offsets
    float* const outb = out + ((size_t)b * ODIM + mofs) * HWDIM
                            + tile0 * COLS_PER_BLOCK + wv * 16 + n16;

    for (int rb_ = 0; rb_ < 4; ++rb_) {        // row-blocks OUTER
        const int arow = rb_ * 16 + n16;

        // A fragments for this row-block: 3 branches x 2 K-chunks,
        // loaded ONCE and reused across all 5 tiles.
        v16bf a[3][2];
        #pragma unroll
        for (int br = 0; br < 3; ++br) {
            #pragma unroll
            for (int kc = 0; kc < 2; ++kc) {
                a[br][kc] = ld_frag(&Wl[br * 4096 + arow * CDIM + kc * 32 + khalf]);
            }
        }

        for (int t = 0; t < TILES_PER_BLOCK; ++t) {   // tiles INNER
            // B fragments for this wave's 16-column slab of tile t
            v16bf bfr[2];
            #pragma unroll
            for (int kc = 0; kc < 2; ++kc) {
                bfr[kc] = ld_frag(&Xs[t][lcol * CDIM + kc * 32 + khalf]);
            }

            v8f acc0 = {}, acc1 = {}, acc2 = {};
            #pragma unroll
            for (int kc = 0; kc < 2; ++kc) {
                acc0 = __builtin_amdgcn_wmma_f32_16x16x32_bf16(false, a[0][kc], false, bfr[kc],
                                                               (short)0, acc0, false, false);
                acc1 = __builtin_amdgcn_wmma_f32_16x16x32_bf16(false, a[1][kc], false, bfr[kc],
                                                               (short)0, acc1, false, false);
                acc2 = __builtin_amdgcn_wmma_f32_16x16x32_bf16(false, a[2][kc], false, bfr[kc],
                                                               (short)0, acc2, false, false);
            }

            // ---- fused epilogue: BN + SiLU per branch, weighted combine ----
            float* outp = outb + (size_t)(rb_ * 16) * HWDIM + t * COLS_PER_BLOCK;
            #pragma unroll
            for (int i = 0; i < 8; ++i) {
                const int ch = rb_ * 16 + mofs + i;
                const float vs = silu(acc0[i] * gmS[ch]       + btS[ch]);
                const float v1 = silu(acc1[i] * gmS[64 + ch]  + btS[64 + ch]);
                const float v2 = silu(acc2[i] * gmS[128 + ch] + btS[128 + ch]);
                outp[(size_t)i * HWDIM] = vs + w0 * v1 + w1 * v2;
            }
        }
    }
}

// ---------------------------------------------------------------------------
// Host-side launcher
// ---------------------------------------------------------------------------
extern "C" void kernel_launch(void* const* d_in, const int* in_sizes, int n_in,
                              void* d_out, int out_size, void* d_ws, size_t ws_size,
                              hipStream_t stream) {
    const float* x            = (const float*)d_in[0];
    const float* router_w     = (const float*)d_in[1];
    const float* router_b     = (const float*)d_in[2];
    const float* shared_w     = (const float*)d_in[3];
    const float* shared_gamma = (const float*)d_in[4];
    const float* shared_beta  = (const float*)d_in[5];
    const float* expert_w     = (const float*)d_in[6];
    const float* expert_gamma = (const float*)d_in[7];
    const float* expert_beta  = (const float*)d_in[8];
    float* out = (float*)d_out;

    int*   idx_r = (int*)d_ws;                          // [B][2]
    float* w_r   = (float*)((char*)d_ws + 512);         // [B][2]

    moe_router_kernel<<<dim3(BATCH), dim3(256), 0, stream>>>(
        x, router_w, router_b, idx_r, w_r);

    dim3 grid(GRID_X, BATCH);                           // (10, 64)
    moe_main_kernel<<<grid, dim3(256), 0, stream>>>(
        x, shared_w, shared_gamma, shared_beta,
        expert_w, expert_gamma, expert_beta,
        idx_r, w_r, out);
}
